// AttentionLayer_36962488549663
// MI455X (gfx1250) — matrix operand order repaired and checked by reading, not compiled, not run on hardware
//
#include <hip/hip_runtime.h>
#include <stddef.h>

#define D_MODEL  1024
#define N_HEADS  16
#define HEAD_DIM 64
#define BATCH    4
#define SEQ      2048

typedef __attribute__((ext_vector_type(16))) __bf16 v16bf;
typedef __attribute__((ext_vector_type(8)))  __bf16 v8bf;
typedef __attribute__((ext_vector_type(4)))  __bf16 v4bf;
typedef __attribute__((ext_vector_type(8)))  float  v8f;
typedef __attribute__((ext_vector_type(4)))  float  f32x4;

static __device__ __forceinline__ v8f wmma_bf16(v16bf a, v16bf b, v8f c) {
  // D = A(16x32 bf16) * B(32x16 bf16) + C(16x16 f32)
  return __builtin_amdgcn_wmma_f32_16x16x32_bf16(false, a, false, b, (short)0, c,
                                                 false, false);
}

static __device__ __forceinline__ v8f zero8() {
  v8f z = {0.f, 0.f, 0.f, 0.f, 0.f, 0.f, 0.f, 0.f};
  return z;
}

// Async global -> LDS copy of 16 bytes per lane (ASYNCcnt-tracked, no VGPR
// round trip). vdst = LDS byte address, vaddr = 64-bit global address.
static __device__ __forceinline__ void async_copy_b128(const void* gptr,
                                                       void* lptr) {
  unsigned int l = (unsigned int)(size_t)(lptr);
  asm volatile("global_load_async_to_lds_b128 %0, %1, off"
               :
               : "v"(l), "v"(gptr)
               : "memory");
}

static __device__ __forceinline__ void wait_async() {
  asm volatile("s_wait_asynccnt 0x0" ::: "memory");
}

// ---------------------------------------------------------------------------
// Elementwise fp32 -> bf16 (vectorized), for activation pre-pack.
// ---------------------------------------------------------------------------
__global__ __launch_bounds__(256) void cvt_f32_bf16(
    const float* __restrict__ x, __bf16* __restrict__ y, int n4) {
  int i = blockIdx.x * 256 + threadIdx.x;
  if (i < n4) {
    f32x4 f = ((const f32x4*)x)[i];
    v4bf h;
#pragma unroll
    for (int j = 0; j < 4; ++j) h[j] = (__bf16)f[j];
    ((v4bf*)y)[i] = h;
  }
}

// ---------------------------------------------------------------------------
// Weight pre-pack: W[K][N] fp32 -> Wt[N][K] bf16 (tiled transpose via LDS).
// ---------------------------------------------------------------------------
__global__ __launch_bounds__(256) void pack_wt(
    const float* __restrict__ W, __bf16* __restrict__ Wt, int K, int N) {
  __shared__ __bf16 t[32][33];
  const int tx = threadIdx.x & 31;
  const int ty = threadIdx.x >> 5;          // 0..7
  const int k0 = blockIdx.y * 32;
  const int n0 = blockIdx.x * 32;
#pragma unroll
  for (int yy = ty; yy < 32; yy += 8)
    t[yy][tx] = (__bf16)W[(size_t)(k0 + yy) * N + n0 + tx];
  __syncthreads();
#pragma unroll
  for (int yy = ty; yy < 32; yy += 8)
    Wt[(size_t)(n0 + yy) * K + k0 + tx] = t[tx][yy];
}

// ---------------------------------------------------------------------------
// C = A[M,K](bf16) * Bt[N,K](bf16, K-major) + bias, bf16 WMMA, f32 accum.
// Block tile 128x128, 256 threads = 8 waves (2x4), wave = 64x32 (4x2 tiles).
// K-step 64, double-buffered LDS fed by async global->LDS b128 copies.
// OUT_MODE 0: C[M][N] row-major.  OUT_MODE 1: per-head transposed
//            C[((b*H + h)*E + e)*SEQ + s]  (for V, so attention can stream
//            V^T tiles with contiguous async copies).
// ---------------------------------------------------------------------------
template <typename OT, int OUT_MODE>
__global__ __launch_bounds__(256) void gemm_bt_wmma(
    const __bf16* __restrict__ A, const __bf16* __restrict__ Bt,
    const float* __restrict__ bias, OT* __restrict__ C,
    int M, int N, int K) {
  __shared__ __bf16 lA[2][128 * 64];   // [m][k]
  __shared__ __bf16 lB[2][128 * 64];   // [n][k]

  const int tid  = threadIdx.x;
  const int lane = tid & 31;
  const int wave = tid >> 5;
  const int mw   = wave >> 2;
  const int nw   = wave & 3;
  const int lm   = lane & 15;
  const int hi   = lane >> 4;
  const int m0   = blockIdx.y * 128;
  const int n0   = blockIdx.x * 128;

  v8f acc[4][2];
#pragma unroll
  for (int ms = 0; ms < 4; ++ms)
#pragma unroll
    for (int ns = 0; ns < 2; ++ns) acc[ms][ns] = zero8();

  auto stage = [&](int kt, int buf) {
#pragma unroll
    for (int it = 0; it < 4; ++it) {
      int i = tid + it * 256;            // 0..1023 chunks of 8 bf16
      int r = i >> 3;                    // 0..127
      int c = (i & 7) * 8;               // 0..56
      async_copy_b128(&A[(size_t)(m0 + r) * K + kt + c], &lA[buf][r * 64 + c]);
      async_copy_b128(&Bt[(size_t)(n0 + r) * K + kt + c], &lB[buf][r * 64 + c]);
    }
  };

  stage(0, 0);
  wait_async();
  __syncthreads();

  int buf = 0;
  for (int kt = 0; kt < K; kt += 64) {
    if (kt + 64 < K) stage(kt + 64, buf ^ 1);

#pragma unroll
    for (int cc = 0; cc < 2; ++cc) {     // two 32-wide K chunks
      const int kb = cc * 32 + hi * 16;  // A-frag: K = kb+{0..15}
      v16bf af[4];
#pragma unroll
      for (int ms = 0; ms < 4; ++ms) {
        int r = mw * 64 + ms * 16 + lm;
        v8bf lo = *(const v8bf*)&lA[buf][r * 64 + kb];
        v8bf hh = *(const v8bf*)&lA[buf][r * 64 + kb + 8];
        v16bf a;
#pragma unroll
        for (int e = 0; e < 8; ++e) { a[e] = lo[e]; a[e + 8] = hh[e]; }
        af[ms] = a;
      }
      v16bf bfr[2];
#pragma unroll
      for (int ns = 0; ns < 2; ++ns) {
        int n = nw * 32 + ns * 16 + lm;  // B-frag: 16 contiguous K values
        bfr[ns] = *(const v16bf*)&lB[buf][n * 64 + cc * 32 + hi * 16];
      }
#pragma unroll
      for (int ms = 0; ms < 4; ++ms)
#pragma unroll
        for (int ns = 0; ns < 2; ++ns)
          acc[ms][ns] = wmma_bf16(af[ms], bfr[ns], acc[ms][ns]);
    }

    wait_async();
    __syncthreads();
    buf ^= 1;
  }

  // ---- epilogue ----
#pragma unroll
  for (int ms = 0; ms < 4; ++ms)
#pragma unroll
    for (int ns = 0; ns < 2; ++ns) {
      int rbase = m0 + mw * 64 + ms * 16 + hi * 8;
      int col   = n0 + nw * 32 + ns * 16 + lm;
      float bsum = bias[col];
#pragma unroll
      for (int v = 0; v < 8; ++v) {
        float val = acc[ms][ns][v] + bsum;
        int row = rbase + v;
        if constexpr (OUT_MODE == 0) {
          C[(size_t)row * N + col] = (OT)val;
        } else {
          // token row -> (b, s); col -> (h, e); store [b][h][e][s]
          int bb = row >> 11, ss = row & (SEQ - 1);
          int h = col >> 6, e = col & (HEAD_DIM - 1);
          C[(((size_t)(bb * N_HEADS + h) * HEAD_DIM + e) * SEQ) + ss] = (OT)val;
        }
      }
    }
}

// ---------------------------------------------------------------------------
// Flash attention: grid (SEQ/64, BATCH*N_HEADS), 128 threads = 4 waves.
// Q: [B*L][D_MODEL] bf16 (token-major).  K: same.  VT: [B][H][E][S] bf16.
// Double-buffered 64-wide K/V tiles streamed via async global->LDS copies.
// ---------------------------------------------------------------------------
__global__ __launch_bounds__(128) void flash_attn_wmma(
    const __bf16* __restrict__ Q, const __bf16* __restrict__ Kb,
    const __bf16* __restrict__ VT, __bf16* __restrict__ ctx) {
  __shared__ __bf16 Kt[2][64 * 64];    // [s][e]
  __shared__ __bf16 Vt[2][64 * 64];    // [e][s]
  __shared__ __bf16 Pl[4][16 * 64];    // per-wave P bounce [m][s]

  const int tid  = threadIdx.x;
  const int lane = tid & 31;
  const int wave = tid >> 5;
  const int lm   = lane & 15;
  const int hi   = lane >> 4;
  const int bh   = blockIdx.y;
  const int b    = bh >> 4;
  const int h    = bh & 15;
  const int q0   = blockIdx.x * 64 + wave * 16;
  const float scale = 0.125f;          // 1/sqrt(64)

  // Q fragments: 2 x (16x32) A-frags over E=64, resident in VGPRs.
  v16bf qf[2];
  {
    const size_t rowbase =
        ((size_t)(b * SEQ + q0 + lm)) * D_MODEL + h * HEAD_DIM;
#pragma unroll
    for (int c = 0; c < 2; ++c) {
      v8bf lo = *(const v8bf*)&Q[rowbase + c * 32 + hi * 16];
      v8bf hh = *(const v8bf*)&Q[rowbase + c * 32 + hi * 16 + 8];
      v16bf a;
#pragma unroll
      for (int e = 0; e < 8; ++e) { a[e] = lo[e]; a[e + 8] = hh[e]; }
      qf[c] = a;
    }
  }

  auto stage = [&](int s0, int buf) {
#pragma unroll
    for (int it = 0; it < 4; ++it) {
      int i = tid + it * 128;          // 0..511 chunks of 8 bf16
      int r = i >> 3;                  // 0..63
      int c = (i & 7) * 8;             // 0..56
      async_copy_b128(&Kb[((size_t)(b * SEQ + s0 + r)) * D_MODEL +
                          h * HEAD_DIM + c],
                      &Kt[buf][r * 64 + c]);
      async_copy_b128(&VT[((size_t)(bh * HEAD_DIM + r)) * SEQ + s0 + c],
                      &Vt[buf][r * 64 + c]);
    }
  };

  v8f o[4];
  float m_[8], l_[8];
#pragma unroll
  for (int n = 0; n < 4; ++n) o[n] = zero8();
#pragma unroll
  for (int v = 0; v < 8; ++v) { m_[v] = -__builtin_inff(); l_[v] = 0.f; }

  stage(0, 0);
  wait_async();
  __syncthreads();

  int buf = 0;
  for (int s0 = 0; s0 < SEQ; s0 += 64) {
    if (s0 + 64 < SEQ) stage(s0 + 64, buf ^ 1);

    // ---- scores: 4 column tiles of 16x16, K-dim = E = 64 ----
    v8f sacc[4];
#pragma unroll
    for (int j = 0; j < 4; ++j) {
      v8f s = zero8();
#pragma unroll
      for (int c = 0; c < 2; ++c) {
        v16bf kf =
            *(const v16bf*)&Kt[buf][(j * 16 + lm) * 64 + c * 32 + hi * 16];
        s = wmma_bf16(qf[c], kf, s);
      }
      sacc[j] = s;
    }

    // ---- online softmax (rows live across 16-lane halves) ----
#pragma unroll
    for (int v = 0; v < 8; ++v) {
#pragma unroll
      for (int j = 0; j < 4; ++j) sacc[j][v] *= scale;
      float x = fmaxf(fmaxf(sacc[0][v], sacc[1][v]),
                      fmaxf(sacc[2][v], sacc[3][v]));
      x = fmaxf(x, __shfl_xor(x, 1, 32));
      x = fmaxf(x, __shfl_xor(x, 2, 32));
      x = fmaxf(x, __shfl_xor(x, 4, 32));
      x = fmaxf(x, __shfl_xor(x, 8, 32));
      float mn   = fmaxf(m_[v], x);
      float corr = __expf(m_[v] - mn);
      m_[v] = mn;
      float rs = 0.f;
#pragma unroll
      for (int j = 0; j < 4; ++j) {
        float p = __expf(sacc[j][v] - mn);
        sacc[j][v] = p;
        rs += p;
      }
      rs += __shfl_xor(rs, 1, 32);
      rs += __shfl_xor(rs, 2, 32);
      rs += __shfl_xor(rs, 4, 32);
      rs += __shfl_xor(rs, 8, 32);
      l_[v] = l_[v] * corr + rs;
#pragma unroll
      for (int n = 0; n < 4; ++n) o[n][v] *= corr;
      int prow = v + hi * 8;
#pragma unroll
      for (int j = 0; j < 4; ++j)
        Pl[wave][prow * 64 + j * 16 + lm] = (__bf16)sacc[j][v];
    }
    // P bounce is wave-private; DS ops from one wave complete in order, so no
    // cross-wave barrier is needed between the stores above and loads below.

    // ---- O += P * V  (K-dim = 64 -> 2 P-frags, 4 e-tiles) ----
#pragma unroll
    for (int kk = 0; kk < 2; ++kk) {
      int kb2 = kk * 32 + hi * 16;
      v8bf lo = *(const v8bf*)&Pl[wave][lm * 64 + kb2];
      v8bf hh = *(const v8bf*)&Pl[wave][lm * 64 + kb2 + 8];
      v16bf pf;
#pragma unroll
      for (int e = 0; e < 8; ++e) { pf[e] = lo[e]; pf[e + 8] = hh[e]; }
#pragma unroll
      for (int n = 0; n < 4; ++n) {
        v16bf vf =
            *(const v16bf*)&Vt[buf][(n * 16 + lm) * 64 + kk * 32 + hi * 16];
        o[n] = wmma_bf16(pf, vf, o[n]);
      }
    }

    wait_async();
    __syncthreads();
    buf ^= 1;
  }

  // ---- normalize + write context (bf16, [B*L][D_MODEL]) ----
  const int rbase = b * SEQ + q0 + hi * 8;
#pragma unroll
  for (int n = 0; n < 4; ++n) {
    int col = h * HEAD_DIM + n * 16 + lm;
#pragma unroll
    for (int v = 0; v < 8; ++v) {
      float val = o[n][v] / l_[v];
      ctx[(size_t)(rbase + v) * D_MODEL + col] = (__bf16)val;
    }
  }
}

// ---------------------------------------------------------------------------
extern "C" void kernel_launch(void* const* d_in, const int* in_sizes, int n_in,
                              void* d_out, int out_size, void* d_ws, size_t ws_size,
                              hipStream_t stream) {
  (void)in_sizes; (void)n_in; (void)out_size; (void)ws_size;
  const float* queries = (const float*)d_in[0];
  const float* keys    = (const float*)d_in[1];
  const float* values  = (const float*)d_in[2];
  const float* Wq = (const float*)d_in[3];
  const float* bq = (const float*)d_in[4];
  const float* Wk = (const float*)d_in[5];
  const float* bk = (const float*)d_in[6];
  const float* Wv = (const float*)d_in[7];
  const float* bv = (const float*)d_in[8];
  const float* Wo = (const float*)d_in[9];
  const float* bo = (const float*)d_in[10];
  float* out = (float*)d_out;

  const int tokens = BATCH * SEQ;           // 8192
  const size_t TD  = (size_t)tokens * D_MODEL;  // 8M elements
  const size_t WD  = (size_t)D_MODEL * D_MODEL; // 1M elements

  __bf16* p   = (__bf16*)d_ws;
  __bf16* Xq  = p; p += TD;                 // bf16 activations
  __bf16* Xk  = p; p += TD;
  __bf16* Xv  = p; p += TD;
  __bf16* WqT = p; p += WD;                 // bf16 weights, [N][K]
  __bf16* WkT = p; p += WD;
  __bf16* WvT = p; p += WD;
  __bf16* WoT = p; p += WD;
  __bf16* Qb  = p; p += TD;                 // projected Q (token-major)
  __bf16* Kb  = p; p += TD;                 // projected K (token-major)
  __bf16* VTp = p; p += TD;                 // projected V, [B][H][E][S]
  __bf16* Ctx = p; p += TD;                 // attention output (token-major)

  // ---- pre-pack (one-time, ~150 MB total traffic @ 23.3 TB/s) ----
  const int n4 = (int)(TD / 4);
  cvt_f32_bf16<<<(n4 + 255) / 256, 256, 0, stream>>>(queries, Xq, n4);
  cvt_f32_bf16<<<(n4 + 255) / 256, 256, 0, stream>>>(keys,    Xk, n4);
  cvt_f32_bf16<<<(n4 + 255) / 256, 256, 0, stream>>>(values,  Xv, n4);
  dim3 gw(D_MODEL / 32, D_MODEL / 32);
  pack_wt<<<gw, 256, 0, stream>>>(Wq, WqT, D_MODEL, D_MODEL);
  pack_wt<<<gw, 256, 0, stream>>>(Wk, WkT, D_MODEL, D_MODEL);
  pack_wt<<<gw, 256, 0, stream>>>(Wv, WvT, D_MODEL, D_MODEL);
  pack_wt<<<gw, 256, 0, stream>>>(Wo, WoT, D_MODEL, D_MODEL);

  // ---- QKV projections (bf16 WMMA, async double-buffered staging) ----
  dim3 gg(D_MODEL / 128, tokens / 128);     // (8, 64)
  gemm_bt_wmma<__bf16, 0><<<gg, 256, 0, stream>>>(
      Xq, WqT, bq, Qb, tokens, D_MODEL, D_MODEL);
  gemm_bt_wmma<__bf16, 0><<<gg, 256, 0, stream>>>(
      Xk, WkT, bk, Kb, tokens, D_MODEL, D_MODEL);
  gemm_bt_wmma<__bf16, 1><<<gg, 256, 0, stream>>>(
      Xv, WvT, bv, VTp, tokens, D_MODEL, D_MODEL);

  // ---- flash attention ----
  flash_attn_wmma<<<dim3(SEQ / 64, BATCH * N_HEADS), 128, 0, stream>>>(
      Qb, Kb, VTp, Ctx);

  // ---- output projection (fp32 out) ----
  gemm_bt_wmma<float, 0><<<gg, 256, 0, stream>>>(
      Ctx, WoT, bo, out, tokens, D_MODEL, D_MODEL);
}
